// QuantBasicTransformerBlock_56478819942880
// MI455X (gfx1250) — matrix operand order repaired and checked by reading, not compiled, not run on hardware
//
#include <hip/hip_runtime.h>
#include <hip/hip_bf16.h>
#include <math.h>

// ---------------------------------------------------------------------------
// Problem constants (from reference)
// ---------------------------------------------------------------------------
#define NTOK   4096     // sequence length of x
#define DIM    320      // model dim
#define HEADS  8
#define DHEAD  40
#define DHP    64       // head dim padded to 2 x 32 for WMMA K-loop
#define CTXLEN 77
#define CTXPAD 80       // 77 padded to multiple of 16 (attention j range)
#define CTXM   128      // context rows padded to multiple of 64 (GEMM M)
#define CTXDIM 768
#define FF2    2560     // 2 * INNER_FF
#define FFI    1280     // INNER_FF

typedef __attribute__((ext_vector_type(16))) _Float16 v16h;
typedef __attribute__((ext_vector_type(8)))  _Float16 v8h;
typedef __attribute__((ext_vector_type(8)))  float    v8f;

static __device__ __forceinline__ v8f wmma_f16(v16h a, v16h b, v8f c) {
  // D = A(16x32 f16) * B(32x16 f16) + C(16x16 f32)
  return __builtin_amdgcn_wmma_f32_16x16x32_f16(false, a, false, b, (short)0, c,
                                                false, false);
}

static __device__ __forceinline__ v16h combine8(v8h lo, v8h hi) {
  v16h r;
#pragma unroll
  for (int i = 0; i < 8; ++i) {
    r[i] = lo[i];
    r[i + 8] = hi[i];
  }
  return r;
}

// A-fragment: 16x32 f16 tile, A[M][K] row-major, fully padded -> unguarded.
// ISA layout: lanes 0-15 (row M=lane):   K halves {k0+0..7, k0+16..23}
//             lanes 16-31 (row M=ln-16): K halves {k0+8..15, k0+24..31}
// => per lane: two contiguous 16-byte loads at p and p+16 halves.
static __device__ __forceinline__ v16h load_A_f16(const _Float16* A, int lda, int m0,
                                                  int k0) {
  int lane = threadIdx.x & 31;
  const _Float16* p =
      A + (size_t)(m0 + (lane & 15)) * lda + k0 + (lane >> 4) * 8;
  v8h lo = *(const v8h*)p;
  v8h hi = *(const v8h*)(p + 16);
  return combine8(lo, hi);
}

// A-fragment from f32 source (LDS P matrix), same layout, converted to f16.
static __device__ __forceinline__ v16h load_A_f32(const float* A, int lda, int m0,
                                                  int k0) {
  int lane = threadIdx.x & 31;
  const float* p = A + (size_t)(m0 + (lane & 15)) * lda + k0 + (lane >> 4) * 8;
  v8f lo = *(const v8f*)p;
  v8f hi = *(const v8f*)(p + 16);
  v16h a;
#pragma unroll
  for (int i = 0; i < 8; ++i) {
    a[i] = (_Float16)lo[i];
    a[i + 8] = (_Float16)hi[i];
  }
  return a;
}

// B-fragment: 32x16 f16 tile from TRANSPOSED storage BT[N][K] (logical
// B[k][n] = BT[n*ldk + k]).  ISA layout: lanes 0-15 column n0+lane K halves
// {k0+0..7, k0+16..23}, lanes 16-31 column n0+lane-16 {k0+8..15, k0+24..31}.
static __device__ __forceinline__ v16h load_BT_f16(const _Float16* BT, int ldk, int k0,
                                                   int n0) {
  int lane = threadIdx.x & 31;
  const _Float16* p =
      BT + (size_t)(n0 + (lane & 15)) * ldk + k0 + (lane >> 4) * 8;
  v8h lo = *(const v8h*)p;
  v8h hi = *(const v8h*)(p + 16);
  return combine8(lo, hi);
}

// ---------------------------------------------------------------------------
// GEMM: C[M,N](f32) = A[M,K](f16) @ BT[N,K](f16)^T (+bias)(+residual)
// 2x2 register blocking: each wave owns a 32x32 output tile (4 accumulators),
// every fragment load feeds two WMMAs.  4 waves / block arranged 2x2 -> 64x64
// per workgroup.  All dims padded (M%64==0, N%64==0, K%32==0): no guards.
// ---------------------------------------------------------------------------
__global__ void gemm_f16_f32(const _Float16* __restrict__ A, const _Float16* __restrict__ BT,
                             float* __restrict__ C, int N, int K,
                             const float* __restrict__ bias,
                             const float* __restrict__ residual) {
  int wave = threadIdx.x >> 5;
  int m0 = blockIdx.y * 64 + (wave >> 1) * 32;
  int n0 = blockIdx.x * 64 + (wave & 1) * 32;
  v8f acc00 = {}, acc01 = {}, acc10 = {}, acc11 = {};
#pragma unroll 2
  for (int k0 = 0; k0 < K; k0 += 32) {
    v16h a0 = load_A_f16(A, K, m0, k0);
    v16h a1 = load_A_f16(A, K, m0 + 16, k0);
    v16h b0 = load_BT_f16(BT, K, k0, n0);
    v16h b1 = load_BT_f16(BT, K, k0, n0 + 16);
    acc00 = wmma_f16(a0, b0, acc00);
    acc01 = wmma_f16(a0, b1, acc01);
    acc10 = wmma_f16(a1, b0, acc10);
    acc11 = wmma_f16(a1, b1, acc11);
  }
  int lane = threadIdx.x & 31;
  int colx = lane & 15;
  int rowx = (lane >> 4) * 8;
#pragma unroll
  for (int t = 0; t < 4; ++t) {
    const v8f& acc = (t == 0) ? acc00 : (t == 1) ? acc01 : (t == 2) ? acc10 : acc11;
    int col = n0 + ((t & 1) ? 16 : 0) + colx;
    size_t base = (size_t)(m0 + ((t >> 1) ? 16 : 0) + rowx) * N + col;
    float badd = bias ? bias[col] : 0.0f;
#pragma unroll
    for (int v = 0; v < 8; ++v) {
      float r = acc[v] + badd;
      if (residual) r += residual[base + (size_t)v * N];
      C[base + (size_t)v * N] = r;
    }
  }
}

// ---------------------------------------------------------------------------
// Attention: one (head, 16-query tile) per workgroup, 256 threads (8 waves).
// S = scale * Q K^T kept in dynamic LDS (16 x jPad f32, ~257KB max on CDNA5),
// exact softmax, then O = P V via WMMA (P converted f32->f16 out of LDS).
// qh: [H][NTOK][DHP] f16 (zero-padded d), kh: [H][jPad][DHP] f16 (zero-padded
// rows & d), vT: [H][DHP][jPad] f16 (d-major, zero-padded).
// ---------------------------------------------------------------------------
__global__ void attn_kernel(const _Float16* __restrict__ qh, const _Float16* __restrict__ kh,
                            const _Float16* __restrict__ vT, _Float16* __restrict__ out,
                            int jLen, int jPad, float scale) {
  extern __shared__ float smem[];
  float* S = smem;                  // [16][jPad]
  float* red = smem + 16 * jPad;    // [16][16]
  float* rowv = red + 256;          // [16]
  const int head = blockIdx.y;
  const int i0 = blockIdx.x * 16;
  const int tid = threadIdx.x;
  const int wave = tid >> 5;
  const int lane = tid & 31;
  const _Float16* Q = qh + (size_t)head * NTOK * DHP;
  const _Float16* K = kh + (size_t)head * jPad * DHP;   // [j][d] == BT form
  const _Float16* V = vT + (size_t)head * DHP * jPad;   // [d][j] == BT form

  // --- Phase 1: S = scale * Q K^T -----------------------------------------
  // Q fragments are loop-invariant per wave: hoist them.
  v16h qa0 = load_A_f16(Q, DHP, i0, 0);
  v16h qa1 = load_A_f16(Q, DHP, i0, 32);
  int jT = jPad >> 4;
  for (int jt = wave; jt < jT; jt += 8) {
    v16h b0 = load_BT_f16(K, DHP, 0, jt * 16);
    v16h b1 = load_BT_f16(K, DHP, 32, jt * 16);
    v8f acc = {};
    acc = wmma_f16(qa0, b0, acc);
    acc = wmma_f16(qa1, b1, acc);
    int col = jt * 16 + (lane & 15);
    int r0 = (lane >> 4) * 8;
#pragma unroll
    for (int v = 0; v < 8; ++v) S[(r0 + v) * jPad + col] = acc[v] * scale;
  }
  __syncthreads();

  // --- Phase 2: softmax over each of the 16 rows (16 threads per row) -----
  int row = tid & 15;
  int sub = tid >> 4;  // 0..15
  float m = -1e30f;
  for (int j = sub; j < jLen; j += 16) m = fmaxf(m, S[row * jPad + j]);
  red[row * 16 + sub] = m;
  __syncthreads();
  if (sub == 0) {
    float mm = -1e30f;
    for (int t = 0; t < 16; ++t) mm = fmaxf(mm, red[row * 16 + t]);
    rowv[row] = mm;
  }
  __syncthreads();
  float mrow = rowv[row];
  float s = 0.0f;
  for (int j = sub; j < jPad; j += 16) {
    float p = (j < jLen) ? __expf(S[row * jPad + j] - mrow) : 0.0f;
    S[row * jPad + j] = p;  // padded columns forced to exactly 0
    s += p;
  }
  red[row * 16 + sub] = s;
  __syncthreads();
  if (sub == 0) {
    float ss = 0.0f;
    for (int t = 0; t < 16; ++t) ss += red[row * 16 + t];
    rowv[row] = ss;
  }
  __syncthreads();
  float inv = 1.0f / rowv[row];
  for (int j = sub; j < jPad; j += 16) S[row * jPad + j] *= inv;
  __syncthreads();

  // --- Phase 3: O = P V (waves 0..3 each own one 16-wide d tile) ----------
  if (wave < 4) {
    int dt = wave;
    v8f acc = {};
#pragma unroll 2
    for (int k0 = 0; k0 < jPad; k0 += 32) {
      v16h a = load_A_f32(S, jPad, 0, k0);          // LDS f32 -> f16 fragments
      v16h b = load_BT_f16(V, jPad, k0, dt * 16);   // V d-major: contiguous
      acc = wmma_f16(a, b, acc);
    }
    int d = dt * 16 + (lane & 15);
    int r0 = i0 + (lane >> 4) * 8;
    if (d < DHEAD) {
#pragma unroll
      for (int v = 0; v < 8; ++v)
        out[(size_t)(r0 + v) * DIM + head * DHEAD + d] = (_Float16)acc[v];
    }
  }
}

// ---------------------------------------------------------------------------
// LayerNorm over DIM=320, one row per block (256 threads), f32 in, f16 out.
// ---------------------------------------------------------------------------
__global__ void layernorm_kernel(const float* __restrict__ x, const float* __restrict__ g,
                                 const float* __restrict__ b, _Float16* __restrict__ y) {
  __shared__ float red[256];
  __shared__ float mu_s, rs_s;
  int row = blockIdx.x;
  int tid = threadIdx.x;
  const float* xr = x + (size_t)row * DIM;
  float v0 = (tid < DIM) ? xr[tid] : 0.0f;
  float v1 = (tid + 256 < DIM) ? xr[tid + 256] : 0.0f;
  red[tid] = v0 + v1;
  __syncthreads();
  for (int s = 128; s > 0; s >>= 1) {
    if (tid < s) red[tid] += red[tid + s];
    __syncthreads();
  }
  if (tid == 0) mu_s = red[0] * (1.0f / DIM);
  __syncthreads();
  float mu = mu_s;
  float d0 = (tid < DIM) ? (v0 - mu) : 0.0f;
  float d1 = (tid + 256 < DIM) ? (v1 - mu) : 0.0f;
  red[tid] = d0 * d0 + d1 * d1;
  __syncthreads();
  for (int s = 128; s > 0; s >>= 1) {
    if (tid < s) red[tid] += red[tid + s];
    __syncthreads();
  }
  if (tid == 0) rs_s = rsqrtf(red[0] * (1.0f / DIM) + 1e-5f);
  __syncthreads();
  float rs = rs_s;
  if (tid < DIM)
    y[(size_t)row * DIM + tid] = (_Float16)((v0 - mu) * rs * g[tid] + b[tid]);
  if (tid + 256 < DIM)
    y[(size_t)row * DIM + tid + 256] =
        (_Float16)((v1 - mu) * rs * g[tid + 256] + b[tid + 256]);
}

// Weight convert f32->f16 with transpose: wT[n*K + k] = w[k*N + n]
__global__ void cast_transpose_f16(const float* __restrict__ w, _Float16* __restrict__ wT,
                                   int K, int N) {
  int i = blockIdx.x * blockDim.x + threadIdx.x;
  if (i >= K * N) return;
  int n = i / K;
  int k = i - n * K;
  wT[(size_t)n * K + k] = (_Float16)w[(size_t)k * N + n];
}

// Cast f32->f16 with row zero-padding: dst is [rowsPad][cols], rows>=validRows = 0
__global__ void cast_pad_rows_f16(const float* __restrict__ src, _Float16* __restrict__ dst,
                                  int validRows, int cols, int total) {
  int i = blockIdx.x * blockDim.x + threadIdx.x;
  if (i >= total) return;
  int r = i / cols;
  dst[i] = (r < validRows) ? (_Float16)src[i] : (_Float16)0.0f;
}

// Repack projection output [rows, 320] f32 -> head-major padded f16
// [HEADS][rowsPad][DHP], zero-filling d in [DHEAD,DHP) and rows in [rows,rowsPad).
__global__ void pack_heads(const float* __restrict__ src, _Float16* __restrict__ dst,
                           int rows, int rowsPad) {
  int i = blockIdx.x * blockDim.x + threadIdx.x;
  int total = HEADS * rowsPad * DHP;
  if (i >= total) return;
  int d = i & (DHP - 1);
  int r = (i / DHP) % rowsPad;
  int h = i / (DHP * rowsPad);
  float v = 0.0f;
  if (r < rows && d < DHEAD) v = src[(size_t)r * DIM + h * DHEAD + d];
  dst[i] = (_Float16)v;
}

// Same but transposed (d-major) for V: dst[h][d][r], zero padded.
__global__ void pack_heads_T(const float* __restrict__ src, _Float16* __restrict__ dst,
                             int rows, int rowsPad) {
  int i = blockIdx.x * blockDim.x + threadIdx.x;
  int total = HEADS * DHP * rowsPad;
  if (i >= total) return;
  int r = i % rowsPad;
  int d = (i / rowsPad) % DHP;
  int h = i / (rowsPad * DHP);
  float v = 0.0f;
  if (r < rows && d < DHEAD) v = src[(size_t)r * DIM + h * DHEAD + d];
  dst[i] = (_Float16)v;
}

// GEGLU: g[r][c] = h[r][c] * gelu_exact(h[r][FFI + c]),  f16 output for next GEMM
__global__ void geglu_kernel(const float* __restrict__ h, _Float16* __restrict__ g, int n) {
  int i = blockIdx.x * blockDim.x + threadIdx.x;
  if (i >= n) return;
  int r = i / FFI;
  int c = i - r * FFI;
  float val = h[(size_t)r * FF2 + c];
  float gate = h[(size_t)r * FF2 + FFI + c];
  float ge = 0.5f * gate * (1.0f + erff(gate * 0.70710678118654752f));
  g[i] = (_Float16)(val * ge);
}

// ---------------------------------------------------------------------------
// Host orchestration
// ---------------------------------------------------------------------------
extern "C" void kernel_launch(void* const* d_in, const int* in_sizes, int n_in,
                              void* d_out, int out_size, void* d_ws, size_t ws_size,
                              hipStream_t stream) {
  const float* x     = (const float*)d_in[0];
  const float* ctx   = (const float*)d_in[1];
  const float* n1_g  = (const float*)d_in[2];
  const float* n1_b  = (const float*)d_in[3];
  const float* a1_wq = (const float*)d_in[4];
  const float* a1_wk = (const float*)d_in[5];
  const float* a1_wv = (const float*)d_in[6];
  const float* a1_wo = (const float*)d_in[7];
  const float* a1_bo = (const float*)d_in[8];
  const float* n2_g  = (const float*)d_in[9];
  const float* n2_b  = (const float*)d_in[10];
  const float* a2_wq = (const float*)d_in[11];
  const float* a2_wk = (const float*)d_in[12];
  const float* a2_wv = (const float*)d_in[13];
  const float* a2_wo = (const float*)d_in[14];
  const float* a2_bo = (const float*)d_in[15];
  const float* n3_g  = (const float*)d_in[16];
  const float* n3_b  = (const float*)d_in[17];
  const float* ff_w1 = (const float*)d_in[18];
  const float* ff_b1 = (const float*)d_in[19];
  const float* ff_w2 = (const float*)d_in[20];
  const float* ff_b2 = (const float*)d_in[21];
  (void)in_sizes; (void)n_in; (void)out_size; (void)ws_size;

  // Workspace carve-out (256B aligned slices)
  char* ws = (char*)d_ws;
  size_t off = 0;
  auto carve = [&](size_t bytes) -> void* {
    void* p = ws + off;
    off += (bytes + 255) & ~(size_t)255;
    return p;
  };
  float*     xA   = (float*)carve((size_t)NTOK * DIM * 4);
  float*     xB   = (float*)carve((size_t)NTOK * DIM * 4);
  _Float16*  lnh  = (_Float16*)carve((size_t)NTOK * DIM * 2);
  float*     tmp  = (float*)carve((size_t)NTOK * FF2 * 4);      // largest GEMM output
  _Float16*  qh   = (_Float16*)carve((size_t)HEADS * NTOK * DHP * 2);
  _Float16*  kh   = (_Float16*)carve((size_t)HEADS * NTOK * DHP * 2);
  _Float16*  vT   = (_Float16*)carve((size_t)HEADS * DHP * NTOK * 2);
  _Float16*  ao   = (_Float16*)carve((size_t)NTOK * DIM * 2);   // attention output
  _Float16*  gh   = (_Float16*)carve((size_t)NTOK * FFI * 2);   // geglu output
  _Float16*  ctxh = (_Float16*)carve((size_t)CTXM * CTXDIM * 2);
  _Float16*  wT   = (_Float16*)carve((size_t)DIM * FF2 * 2);    // largest weight (f16, T)

  const float scale = 0.15811388300841897f;  // 40^-0.5

  auto convT = [&](const float* src, _Float16* dst, int K, int N) {
    int n = K * N;
    cast_transpose_f16<<<(n + 255) / 256, 256, 0, stream>>>(src, dst, K, N);
  };
  auto gemm = [&](const _Float16* A, const _Float16* BT, float* C, int M, int N, int K,
                  const float* bias, const float* residual) {
    dim3 grid(N / 64, M / 64);
    gemm_f16_f32<<<grid, 128, 0, stream>>>(A, BT, C, N, K, bias, residual);
  };
  auto pack = [&](const float* src, _Float16* dst, int rows, int rowsPad) {
    int total = HEADS * rowsPad * DHP;
    pack_heads<<<(total + 255) / 256, 256, 0, stream>>>(src, dst, rows, rowsPad);
  };
  auto packT = [&](const float* src, _Float16* dst, int rows, int rowsPad) {
    int total = HEADS * DHP * rowsPad;
    pack_heads_T<<<(total + 255) / 256, 256, 0, stream>>>(src, dst, rows, rowsPad);
  };

  // Residual seed: xA = x
  hipMemcpyAsync(xA, x, (size_t)NTOK * DIM * 4, hipMemcpyDeviceToDevice, stream);

  // ---------------- Self-attention ----------------
  layernorm_kernel<<<NTOK, 256, 0, stream>>>(xA, n1_g, n1_b, lnh);
  convT(a1_wq, wT, DIM, DIM);
  gemm(lnh, wT, tmp, NTOK, DIM, DIM, nullptr, nullptr);
  pack(tmp, qh, NTOK, NTOK);
  convT(a1_wk, wT, DIM, DIM);
  gemm(lnh, wT, tmp, NTOK, DIM, DIM, nullptr, nullptr);
  pack(tmp, kh, NTOK, NTOK);
  convT(a1_wv, wT, DIM, DIM);
  gemm(lnh, wT, tmp, NTOK, DIM, DIM, nullptr, nullptr);
  packT(tmp, vT, NTOK, NTOK);
  {
    size_t lds = (size_t)(16 * NTOK + 256 + 16) * 4;  // ~257 KB (CDNA5: 320KB/WG)
    attn_kernel<<<dim3(NTOK / 16, HEADS), 256, lds, stream>>>(qh, kh, vT, ao, NTOK,
                                                              NTOK, scale);
  }
  convT(a1_wo, wT, DIM, DIM);
  gemm(ao, wT, xB, NTOK, DIM, DIM, a1_bo, xA);  // xB = attn @ wo + bo + xA

  // ---------------- Cross-attention ----------------
  layernorm_kernel<<<NTOK, 256, 0, stream>>>(xB, n2_g, n2_b, lnh);
  convT(a2_wq, wT, DIM, DIM);
  gemm(lnh, wT, tmp, NTOK, DIM, DIM, nullptr, nullptr);
  pack(tmp, qh, NTOK, NTOK);
  {
    int total = CTXM * CTXDIM;
    cast_pad_rows_f16<<<(total + 255) / 256, 256, 0, stream>>>(ctx, ctxh, CTXLEN,
                                                               CTXDIM, total);
  }
  convT(a2_wk, wT, CTXDIM, DIM);
  gemm(ctxh, wT, tmp, CTXM, DIM, CTXDIM, nullptr, nullptr);
  pack(tmp, kh, CTXPAD, CTXPAD);   // rows 77..79 are exact zeros already
  convT(a2_wv, wT, CTXDIM, DIM);
  gemm(ctxh, wT, tmp, CTXM, DIM, CTXDIM, nullptr, nullptr);
  packT(tmp, vT, CTXPAD, CTXPAD);
  {
    size_t lds = (size_t)(16 * CTXPAD + 256 + 16) * 4;  // ~6 KB
    attn_kernel<<<dim3(NTOK / 16, HEADS), 256, lds, stream>>>(qh, kh, vT, ao, CTXLEN,
                                                              CTXPAD, scale);
  }
  convT(a2_wo, wT, DIM, DIM);
  gemm(ao, wT, xA, NTOK, DIM, DIM, a2_bo, xB);  // xA = attn @ wo + bo + xB

  // ---------------- GEGLU feed-forward ----------------
  layernorm_kernel<<<NTOK, 256, 0, stream>>>(xA, n3_g, n3_b, lnh);
  convT(ff_w1, wT, DIM, FF2);
  gemm(lnh, wT, tmp, NTOK, FF2, DIM, ff_b1, nullptr);
  {
    int n = NTOK * FFI;
    geglu_kernel<<<(n + 255) / 256, 256, 0, stream>>>(tmp, gh, n);
  }
  convT(ff_w2, wT, FFI, DIM);
  gemm(gh, wT, (float*)d_out, NTOK, DIM, FFI, ff_b2, xA);  // out = g @ w2 + b2 + xA
}
